// InterectionBlock_33208687133091
// MI455X (gfx1250) — matrix operand order repaired and checked by reading, compiled, not run on hardware
//
#include <hip/hip_runtime.h>
#include <hip/hip_bf16.h>
#include <math.h>

// ---------------------------------------------------------------------------
// CDNA5 (gfx1250) wave32 WMMA implementation of the interaction block.
//   v_wmma_f32_16x16x32_f16 : D(16x16,f32) = A(16x32,f16) x B(32x16,f16) + C
// B operands staged into LDS with GLOBAL_LOAD_ASYNC_TO_LDS_B128 (ASYNCcnt,
// double-buffered k-slabs). 8 waves/block; each wave owns a 16x128 output slab
// (8 acc tiles = 64 VGPRs) so there are no spills and plenty of waves resident
// to hide ds/WMMA latency. Bias is applied in the epilogue so accumulators
// start at (inline) zero.
// Fragment layouts (ISA 7.12.2, wave32):
//   A: lane<16 -> row M=lane,    halfs = K{k0..k0+7, k0+16..k0+23}
//      lane>=16-> row M=lane-16, halfs = K{k0+8..k0+15, k0+24..k0+31}
//   B: lane<16 -> col N=lane,    halfs = K{k0..k0+7, k0+16..k0+23}
//      lane>=16-> col N=lane-16, halfs = K{k0+8..k0+15, k0+24..k0+31}
//   C/D: VGPR r -> M = r (lane<16) / 8+r (lane>=16); N = lane&15
// ---------------------------------------------------------------------------

typedef __attribute__((ext_vector_type(16))) _Float16 v16h;
typedef __attribute__((ext_vector_type(8)))  float    v8f;

#define HF 256            // hidden width (H == F == 256)
#define GEA 50            // edge_attr feature count
#define LOG2F_C 0.69314718055994531f

__device__ __forceinline__ float sspf(float v) {
    float sp = (v > 20.0f) ? v : __logf(1.0f + __expf(v));
    return sp - LOG2F_C;
}

// ---- CDNA5 async global->LDS copy (ASYNCcnt) ------------------------------
__device__ __forceinline__ void async_b128(unsigned lds_off, unsigned long long gaddr) {
    asm volatile("global_load_async_to_lds_b128 %0, %1, off"
                 :: "v"(lds_off), "v"(gaddr) : "memory");
}

// copy one 16 KB B k-slab with 256 threads: 4 async b128 per thread (+4 ASYNCcnt/wave)
__device__ __forceinline__ void stage_slab_256t(const _Float16* gsrc, _Float16* ldst, int tid) {
    unsigned lbase = (unsigned)(uintptr_t)ldst;           // low 32 bits = LDS offset
    unsigned long long g = (unsigned long long)(uintptr_t)gsrc;
#pragma unroll
    for (int i = 0; i < 4; ++i) {
        unsigned off = (unsigned)tid * 16u + (unsigned)i * 4096u;
        async_b128(lbase + off, g + off);
    }
}

template <int N>
__device__ __forceinline__ void wait_asynccnt() {
    asm volatile("s_wait_asynccnt %0" :: "n"(N) : "memory");
}

// ---- weight packing into f16 B-fragment layout ----------------------------
//   offset(k,n) = (k>>5)*8192 + (n>>4)*512 + ((n&15) + (((k>>3)&1)<<4))*16 + ((k&7) | (((k>>4)&1)<<3))
__global__ void pack_b_f16(const float* __restrict__ W, _Float16* __restrict__ out,
                           int Ksrc, int Kpad) {
    int idx = blockIdx.x * blockDim.x + threadIdx.x;
    if (idx >= Kpad * HF) return;
    int k = idx >> 8;
    int n = idx & 255;
    float v = (k < Ksrc) ? W[k * HF + n] : 0.0f;
    int ks  = k >> 5;
    int kk  = k & 31;
    int t   = n >> 4;
    int ln  = (n & 15) | (((kk >> 3) & 1) << 4);
    int pos = (kk & 7) | (((kk >> 4) & 1) << 3);
    out[ks * 8192 + t * 512 + ln * 16 + pos] = (_Float16)v;
}

__global__ void zero_f32(float* __restrict__ p, int n) {
    int i = blockIdx.x * blockDim.x + threadIdx.x;
    if (i < n) p[i] = 0.0f;
}

// A raw loads for one k-step: 4x float4 per lane (K runs {k0+g*8..+7, k0+16+g*8..+7})
__device__ __forceinline__ void load_a_raw(const float* ap, bool valid, int k0, int g,
                                           float4& r0, float4& r1, float4& r2, float4& r3) {
    if (valid) {
        const float* b = ap + k0 + g * 8;
        r0 = *(const float4*)(b + 0);
        r1 = *(const float4*)(b + 4);
        r2 = *(const float4*)(b + 16);
        r3 = *(const float4*)(b + 20);
    } else {
        r0 = r1 = r2 = r3 = (float4){0.f, 0.f, 0.f, 0.f};
    }
}

__device__ __forceinline__ v16h pack_a(const float4& r0, const float4& r1,
                                       const float4& r2, const float4& r3) {
    v16h a;
    a[0]=(_Float16)r0.x; a[1]=(_Float16)r0.y; a[2]=(_Float16)r0.z; a[3]=(_Float16)r0.w;
    a[4]=(_Float16)r1.x; a[5]=(_Float16)r1.y; a[6]=(_Float16)r1.z; a[7]=(_Float16)r1.w;
    a[8]=(_Float16)r2.x; a[9]=(_Float16)r2.y; a[10]=(_Float16)r2.z; a[11]=(_Float16)r2.w;
    a[12]=(_Float16)r3.x; a[13]=(_Float16)r3.y; a[14]=(_Float16)r3.z; a[15]=(_Float16)r3.w;
    return a;
}

// ---------------------------------------------------------------------------
// Node GEMM: out[M,256] = act( A[M,256] @ Bpack + bias ) (+ resid)
// 256 threads = 8 waves; wave (wrow,wcol) owns rows [blk*64+wrow*16,+16) x
// cols [wcol*128,+128) -> 8 acc tiles. B slabs double-buffered in LDS.
// ---------------------------------------------------------------------------
template <bool ACT, bool RESID>
__global__ void gemm_node(const float* __restrict__ A, const _Float16* __restrict__ Bpack,
                          const float* __restrict__ bias, const float* __restrict__ resid,
                          float* __restrict__ out, int M) {
    __shared__ __align__(32) _Float16 Bs[2][8192];

    const int tid  = threadIdx.x;
    const int wave = tid >> 5;
    const int lane = tid & 31;
    const int mloc = lane & 15;
    const int g    = lane >> 4;
    const int wrow = wave >> 1;          // 0..3
    const int wcol = wave & 1;           // 0..1 (N-half)
    const int row0 = blockIdx.x * 64 + wrow * 16;
    const int arow = row0 + mloc;
    const bool aval = (arow < M);
    const float* ap = A + (size_t)arow * HF;

    stage_slab_256t(Bpack, Bs[0], tid);       // slab 0 in flight ASAP

    v8f acc[8];
#pragma unroll
    for (int t = 0; t < 8; ++t) acc[t] = (v8f){0.f,0.f,0.f,0.f,0.f,0.f,0.f,0.f};

    float4 c0, c1, c2, c3;                    // current A raw
    load_a_raw(ap, aval, 0, g, c0, c1, c2, c3);

    for (int ks = 0; ks < 8; ++ks) {
        float4 n0, n1, n2, n3;
        if (ks < 7) {
            stage_slab_256t(Bpack + (ks + 1) * 8192, Bs[(ks + 1) & 1], tid);
            load_a_raw(ap, aval, (ks + 1) * 32, g, n0, n1, n2, n3);
            wait_asynccnt<4>();               // slab ks resident; slab ks+1 may fly
        } else {
            wait_asynccnt<0>();
        }
        __syncthreads();

        v16h a = pack_a(c0, c1, c2, c3);
        const _Float16* bp = &Bs[ks & 1][(wcol * 8) * 512 + lane * 16];
#pragma unroll
        for (int t = 0; t < 8; ++t) {
            v16h b = *(const v16h*)(bp + t * 512);
            acc[t] = __builtin_amdgcn_wmma_f32_16x16x32_f16(
                false, a, false, b, (short)0, acc[t], false, false);
        }
        __syncthreads();                      // all waves done with Bs[ks&1]
        c0 = n0; c1 = n1; c2 = n2; c3 = n3;
    }

#pragma unroll
    for (int t = 0; t < 8; ++t) {
        const int n  = (wcol * 8 + t) * 16 + mloc;
        const float bv = bias ? bias[n] : 0.0f;
#pragma unroll
        for (int r = 0; r < 8; ++r) {
            const int rr = row0 + g * 8 + r;  // C-layout row
            if (rr < M) {
                float v = acc[t][r] + bv;
                if (ACT)   v = sspf(v);
                if (RESID) v += resid[(size_t)rr * HF + n];
                out[(size_t)rr * HF + n] = v;
            }
        }
    }
}

// ---------------------------------------------------------------------------
// Edge kernel: 256 threads = 8 waves; wave (wrow,wcol) handles the 16 edges of
// group wrow and output cols [wcol*128,+128).
//   hidden = ssp(edge_attr @ W1 + b1)   (K=64 padded, B via LDS)
//   Wf     = hidden @ W2 + b2           (K=256, B async-LDS double-buffered)
//   msg    = Wf * C[e] * xl[j];  atomicAdd(agg[i], msg)
// LDS (64 KB), region-aliased:
//   smem[0     .. 16383] : W1 packed (32 KB) during GEMM1, then hid A-fragments
//   smem[16384 .. 32767] : W2 k-slab double buffer (2 x 16 KB)
// ---------------------------------------------------------------------------
__global__ void edge_kernel(const float* __restrict__ edge_attr,
                            const float* __restrict__ edge_w,
                            const long long* __restrict__ edge_index,
                            const _Float16* __restrict__ w1p,
                            const _Float16* __restrict__ w2p,
                            const float* __restrict__ b1,
                            const float* __restrict__ b2,
                            const float* __restrict__ xl,
                            float* __restrict__ agg,
                            int E) {
    __shared__ __align__(32) _Float16 smem[32768];   // 64 KB

    _Float16* w1s = &smem[0];                        // 2 slabs of W1 (32 KB)
    _Float16* hid = &smem[0];                        // recycled after GEMM1
    _Float16* Bs0 = &smem[16384];
    _Float16* Bs1 = &smem[16384 + 8192];

    const int tid   = threadIdx.x;
    const int wave  = tid >> 5;
    const int lane  = tid & 31;
    const int mloc  = lane & 15;
    const int g     = lane >> 4;
    const int wrow  = wave >> 1;         // edge group 0..3
    const int wcol  = wave & 1;          // N-half
    const int ebase = blockIdx.x * 64 + wrow * 16;

    // stage W1 (both slabs) + W2 slab 0 up front: 12 async ops per wave.
    // Async completions are in-order per wave, so asynccnt<=4 => W1 is resident.
    stage_slab_256t(w1p,        w1s,        tid);
    stage_slab_256t(w1p + 8192, w1s + 8192, tid);
    stage_slab_256t(w2p,        Bs0,        tid);

    // --- cutoff envelope, edge (ebase+mloc); overlaps the staging ---
    float cme = 0.0f;
    {
        const int e = ebase + mloc;
        if (e < E) {
            float w0 = edge_w[(size_t)e * 3 + 0];
            float w1 = edge_w[(size_t)e * 3 + 1];
            float w2 = edge_w[(size_t)e * 3 + 2];
            float d  = __fsqrt_rn(w0 * w0 + w1 * w1 + w2 * w2);
            if (d <= 2.0f)                  // 0.5*(cos(d*pi/CUTOFF)+1), CUTOFF=2
                cme = 0.5f * (__cosf(d * 1.57079632679f) + 1.0f);
        }
    }

    // --- A fragments for GEMM1 from edge_attr (issued before the async wait) ---
    const int ea = ebase + mloc;            // A-fragment edge row
    v16h a1[2];
#pragma unroll
    for (int ks = 0; ks < 2; ++ks) {
        const int k0 = ks * 32;
#pragma unroll
        for (int h = 0; h < 8; ++h) {
            int k1 = k0 + g * 8 + h;
            int k2 = k0 + 16 + g * 8 + h;
            a1[ks][h]     = (ea < E && k1 < GEA) ? (_Float16)edge_attr[(size_t)ea * GEA + k1]
                                                 : (_Float16)0.0f;
            a1[ks][8 + h] = (ea < E && k2 < GEA) ? (_Float16)edge_attr[(size_t)ea * GEA + k2]
                                                 : (_Float16)0.0f;
        }
    }

    // --- GEMM 1: hidden = ssp(edge_attr @ W1 + b1), B fragments from LDS ---
    v8f acc[8];
#pragma unroll
    for (int t = 0; t < 8; ++t) acc[t] = (v8f){0.f,0.f,0.f,0.f,0.f,0.f,0.f,0.f};

    wait_asynccnt<4>();                     // W1 resident (W2 slab 0 may still fly)
    __syncthreads();

#pragma unroll
    for (int ks = 0; ks < 2; ++ks) {
        const _Float16* bp = w1s + ks * 8192 + (wcol * 8) * 512 + lane * 16;
#pragma unroll
        for (int t = 0; t < 8; ++t) {
            v16h b = *(const v16h*)(bp + t * 512);
            acc[t] = __builtin_amdgcn_wmma_f32_16x16x32_f16(
                false, a1[ks], false, b, (short)0, acc[t], false, false);
        }
    }
    __syncthreads();                        // all waves done reading W1 region

    // --- activation + bias, store hidden into recycled region (A-frag layout) ---
#pragma unroll
    for (int t = 0; t < 8; ++t) {
        const int n    = (wcol * 8 + t) * 16 + mloc;  // hidden feature == next K
        const float bv = b1[n];
        const int ks2  = n >> 5;
        const int kk   = n & 31;
        const int hi   = (kk & 7) | (((kk >> 4) & 1) << 3);
        const int loff = ((kk >> 3) & 1) << 4;
#pragma unroll
        for (int r = 0; r < 8; ++r) {
            const int m = g * 8 + r;        // C-layout row = edge-in-wave
            hid[wrow * 4096 + ks2 * 512 + (m + loff) * 16 + hi] =
                (_Float16)sspf(acc[t][r] + bv);
        }
    }

    // --- GEMM 2: Wf = hidden @ W2 + b2, B slabs async double-buffered ---
#pragma unroll
    for (int t = 0; t < 8; ++t) acc[t] = (v8f){0.f,0.f,0.f,0.f,0.f,0.f,0.f,0.f};

    for (int ks = 0; ks < 8; ++ks) {
        if (ks < 7) {
            stage_slab_256t(w2p + (ks + 1) * 8192, (ks & 1) ? Bs0 : Bs1, tid);
            wait_asynccnt<4>();             // slab ks resident; slab ks+1 may fly
        } else {
            wait_asynccnt<0>();
        }
        __syncthreads();

        v16h a = *(const v16h*)&hid[wrow * 4096 + ks * 512 + lane * 16];
        const _Float16* bp = ((ks & 1) ? Bs1 : Bs0) + (wcol * 8) * 512 + lane * 16;
#pragma unroll
        for (int t = 0; t < 8; ++t) {
            v16h b = *(const v16h*)(bp + t * 512);
            acc[t] = __builtin_amdgcn_wmma_f32_16x16x32_f16(
                false, a, false, b, (short)0, acc[t], false, false);
        }
        __syncthreads();
    }

    // --- bias, scale by C, gather xl[j], scatter-add into agg[i] ---
    long long iv[8], jv[8];
    float     cv[8];
    bool      ok[8];
#pragma unroll
    for (int r = 0; r < 8; ++r) {
        const int e = ebase + g * 8 + r;    // C-layout row -> edge id
        ok[r] = (e < E);
        iv[r] = ok[r] ? edge_index[e]     : 0;   // destination i
        jv[r] = ok[r] ? edge_index[E + e] : 0;   // source j
        cv[r] = __shfl(cme, g * 8 + r, 32);
    }
#pragma unroll
    for (int t = 0; t < 8; ++t) {
        const int n  = (wcol * 8 + t) * 16 + mloc;
        const float bv = b2[n];
#pragma unroll
        for (int r = 0; r < 8; ++r) {
            if (ok[r]) {
                float v = (acc[t][r] + bv) * cv[r] * xl[(size_t)jv[r] * HF + n];
                atomicAdd(&agg[(size_t)iv[r] * HF + n], v);
            }
        }
    }
}

// ---------------------------------------------------------------------------
extern "C" void kernel_launch(void* const* d_in, const int* in_sizes, int n_in,
                              void* d_out, int out_size, void* d_ws, size_t ws_size,
                              hipStream_t stream) {
    const float*     x     = (const float*)d_in[0];
    const long long* eidx  = (const long long*)d_in[1];
    const float*     ew    = (const float*)d_in[2];
    const float*     eattr = (const float*)d_in[3];
    const float*     w1    = (const float*)d_in[4];
    const float*     b1    = (const float*)d_in[5];
    const float*     w2    = (const float*)d_in[6];
    const float*     b2    = (const float*)d_in[7];
    const float*     l1w   = (const float*)d_in[8];
    const float*     l2w   = (const float*)d_in[9];
    const float*     l2b   = (const float*)d_in[10];
    float*           out   = (float*)d_out;

    const int N = in_sizes[0] / HF;   // 10000
    const int E = in_sizes[2] / 3;    // 320000

    // workspace layout
    float*    xl   = (float*)d_ws;                       // N*256 f32
    float*    agg  = xl + (size_t)N * HF;                // N*256 f32
    _Float16* w1p  = (_Float16*)(agg + (size_t)N * HF);  // 64*256 f16
    _Float16* w2p  = w1p + 64 * HF;                      // 256*256 f16
    _Float16* l1p  = w2p + HF * HF;
    _Float16* l2p  = l1p + HF * HF;

    // 1) pack weights into f16 B-fragment layout
    pack_b_f16<<<(64 * HF + 255) / 256, 256, 0, stream>>>(w1, w1p, GEA, 64);
    pack_b_f16<<<(HF * HF + 255) / 256, 256, 0, stream>>>(w2, w2p, HF, HF);
    pack_b_f16<<<(HF * HF + 255) / 256, 256, 0, stream>>>(l1w, l1p, HF, HF);
    pack_b_f16<<<(HF * HF + 255) / 256, 256, 0, stream>>>(l2w, l2p, HF, HF);

    // 2) zero the segment-sum accumulator
    zero_f32<<<((size_t)N * HF + 255) / 256, 256, 0, stream>>>(agg, N * HF);

    // 3) xl = x @ lin1_w
    gemm_node<false, false><<<(N + 63) / 64, 256, 0, stream>>>(
        x, l1p, nullptr, nullptr, xl, N);

    // 4) edge MLP + envelope + gather + scatter-add
    edge_kernel<<<(E + 63) / 64, 256, 0, stream>>>(
        eattr, ew, eidx, w1p, w2p, b1, b2, xl, agg, E);

    // 5) out = x + ssp(agg @ lin2_w + lin2_b)
    gemm_node<true, true><<<(N + 63) / 64, 256, 0, stream>>>(
        agg, l2p, l2b, x, out, N);
}
